// TRENDSPOT_74500502717030
// MI455X (gfx1250) — matrix operand order repaired and checked by reading, not compiled, not run on hardware
//
#include <hip/hip_runtime.h>
#include <hip/hip_bf16.h>
#include <hip/hip_fp16.h>

typedef __attribute__((ext_vector_type(16))) _Float16 v16h;
typedef __attribute__((ext_vector_type(8)))  _Float16 v8h;
typedef __attribute__((ext_vector_type(8)))  float    v8f;

#define LAGT 50

__device__ __forceinline__ float sigmoidf_(float x) { return 1.f / (1.f + __expf(-x)); }

// branch-free tanh on the native-exp path: e = exp(-2|x|) in (0,1]
__device__ __forceinline__ float tanhf_(float x) {
    const float e = __expf(-2.f * __builtin_fabsf(x));
    const float t = (1.f - e) / (1.f + e);
    return __builtin_copysignf(t, x);
}

// Build A fragment (16x32 f16, contiguous-K layout, k = 16*hf + s) from an LDS
// row-major [16][32] f16 tile.  lane: hf = lane>>4, n = lane&15, row m = n.
__device__ __forceinline__ v16h loadA16x32(const _Float16* lds_h, int n, int hf) {
    const v8h* plo = (const v8h*)(lds_h + n * 32 + hf * 16);      // K = hf*16 .. +7
    const v8h* phi = (const v8h*)(lds_h + n * 32 + hf * 16 + 8);  // K = hf*16+8 .. +7
    v8h lo = *plo, hi = *phi;
    v16h A;
#pragma unroll
    for (int i = 0; i < 8; i++) { A[i] = lo[i]; A[8 + i] = hi[i]; }
    return A;
}

// ---------------------------------------------------------------------------
// Kernel 1: fused LSTM (50 steps) + online-softmax attention pooling + FC+relu
// One wave (32 lanes) owns a 16-node tile. 8 WMMAs/step compute all 128 gates.
// ---------------------------------------------------------------------------
__global__ __launch_bounds__(32) void lstm_att_fc(
    const float* __restrict__ yx,    // [N,50]
    const float* __restrict__ w_ih,  // [128]
    const float* __restrict__ w_hh,  // [128,32]
    const float* __restrict__ b_ih,  // [128]
    const float* __restrict__ b_hh,  // [128]
    const float* __restrict__ attW,  // [50,32]
    const float* __restrict__ fcW,   // [32,32]
    const float* __restrict__ fcb,   // [32]
    float* __restrict__ x1, int N)
{
    __shared__ __attribute__((aligned(16))) _Float16 lds_h[16 * 32];  // h tile f16
    __shared__ float lds_x[LAGT * 16];                                // yx tile [t][m]

    const int lane = threadIdx.x;
    const int hf   = lane >> 4;    // lane half
    const int n    = lane & 15;    // column within tile
    int base = blockIdx.x * 16;
    if (base + 16 > N) base = N - 16;   // clamp (duplicate work, deterministic)

    // preload node_yx tile transposed into LDS
    for (int idx = lane; idx < 16 * LAGT; idx += 32) {
        int m = idx / LAGT, t = idx % LAGT;
        lds_x[t * 16 + m] = yx[(size_t)(base + m) * LAGT + t];
    }

    // w_hh B fragments: B[k][g] = w_hh[g][k]; B 32x16: lanes0-15 K=0..15, lanes16-31 K=16..31
    v16h Bw[8];
#pragma unroll
    for (int j = 0; j < 8; j++) {
        const int g = j * 16 + n;
#pragma unroll
        for (int i = 0; i < 16; i++)
            Bw[j][i] = (_Float16)w_hh[g * 32 + hf * 16 + i];
    }
    // input-projection constants (one gate column per lane per fragment)
    float wih[8], bias[8];
#pragma unroll
    for (int j = 0; j < 8; j++) {
        const int g = j * 16 + n;
        wih[j]  = w_ih[g];
        bias[j] = b_ih[g] + b_hh[g];
    }

    // register state: node m = v + 8*hf ; hidden k0 = n, k1 = 16+n
    float h0[8], h1[8], c0[8], c1[8], P0[8], P1[8], Mx[8], Sx[8];
#pragma unroll
    for (int v = 0; v < 8; v++) {
        h0[v] = h1[v] = c0[v] = c1[v] = P0[v] = P1[v] = 0.f;
        Sx[v] = 0.f; Mx[v] = -1e30f;
    }

    for (int t = 0; t < LAGT; t++) {
        // store current h as f16 into LDS (C/D layout -> row-major tile)
#pragma unroll
        for (int v = 0; v < 8; v++) {
            const int m = v + hf * 8;
            lds_h[m * 32 + n]      = (_Float16)h0[v];
            lds_h[m * 32 + 16 + n] = (_Float16)h1[v];
        }
        __syncthreads();  // single wave: forces DS waits; DS is in-order per wave

        const v16h A = loadA16x32(lds_h, n, hf);

        float xs[8];
#pragma unroll
        for (int v = 0; v < 8; v++) xs[v] = lds_x[t * 16 + v + hf * 8];

        // gates = xp + h @ w_hh^T + b   via 8 WMMAs (C carries the input proj)
        v8f D[8];
#pragma unroll
        for (int j = 0; j < 8; j++) {
            v8f C;
#pragma unroll
            for (int v = 0; v < 8; v++) C[v] = xs[v] * wih[j] + bias[j];
            D[j] = __builtin_amdgcn_wmma_f32_16x16x32_f16(
                false, A, false, Bw[j], (short)0, C, false, false);
        }

        const float aw0 = attW[t * 32 + n];
        const float aw1 = attW[t * 32 + 16 + n];
#pragma unroll
        for (int v = 0; v < 8; v++) {
            const float i0 = sigmoidf_(D[0][v]), i1 = sigmoidf_(D[1][v]);
            const float f0 = sigmoidf_(D[2][v]), f1 = sigmoidf_(D[3][v]);
            const float g0 = tanhf_(D[4][v]),    g1 = tanhf_(D[5][v]);
            const float o0 = sigmoidf_(D[6][v]), o1 = sigmoidf_(D[7][v]);
            c0[v] = f0 * c0[v] + i0 * g0;
            c1[v] = f1 * c1[v] + i1 * g1;
            h0[v] = o0 * tanhf_(c0[v]);
            h1[v] = o1 * tanhf_(c1[v]);
            // attention score: reduce over k across the 16-lane half
            float part = h0[v] * aw0 + h1[v] * aw1;
            part += __shfl_xor(part, 1);
            part += __shfl_xor(part, 2);
            part += __shfl_xor(part, 4);
            part += __shfl_xor(part, 8);
            // online softmax accumulation of pooled = sum_t att * h_t
            const float newM = fmaxf(Mx[v], part);
            const float sc = __expf(Mx[v] - newM);
            const float w  = __expf(part - newM);
            Sx[v] = Sx[v] * sc + w;
            P0[v] = P0[v] * sc + w * h0[v];
            P1[v] = P1[v] * sc + w * h1[v];
            Mx[v] = newM;
        }
        __syncthreads();
    }

    // pooled -> LDS (f16), then x1 = relu(pooled @ fcW + fcb) via 2 WMMAs
#pragma unroll
    for (int v = 0; v < 8; v++) {
        const float inv = 1.0f / Sx[v];
        const int m = v + hf * 8;
        lds_h[m * 32 + n]      = (_Float16)(P0[v] * inv);
        lds_h[m * 32 + 16 + n] = (_Float16)(P1[v] * inv);
    }
    __syncthreads();
    const v16h Ap = loadA16x32(lds_h, n, hf);
#pragma unroll
    for (int j = 0; j < 2; j++) {
        v16h Bf;
#pragma unroll
        for (int i = 0; i < 16; i++)
            Bf[i] = (_Float16)fcW[(hf * 16 + i) * 32 + j * 16 + n];
        v8f C;
        const float bv = fcb[j * 16 + n];
#pragma unroll
        for (int v = 0; v < 8; v++) C[v] = bv;
        v8f Dv = __builtin_amdgcn_wmma_f32_16x16x32_f16(
            false, Ap, false, Bf, (short)0, C, false, false);
#pragma unroll
        for (int v = 0; v < 8; v++)
            x1[(size_t)(base + v + hf * 8) * 32 + j * 16 + n] = fmaxf(Dv[v], 0.f);
    }
}

// ---------------------------------------------------------------------------
// [N,32] @ [32,32] GEMM via WMMA (wave per 16-row tile)
// ---------------------------------------------------------------------------
__global__ __launch_bounds__(32) void gemm_nx32(
    const float* __restrict__ X, const float* __restrict__ W,
    float* __restrict__ Y, int N)
{
    const int lane = threadIdx.x;
    const int hf = lane >> 4, n = lane & 15;
    int base = blockIdx.x * 16;
    if (base + 16 > N) base = N - 16;

    const size_t row = (size_t)(base + n) * 32;
    v16h A;
#pragma unroll
    for (int i = 0; i < 8; i++) {
        A[i]     = (_Float16)X[row + hf * 16 + i];
        A[8 + i] = (_Float16)X[row + hf * 16 + 8 + i];
    }
#pragma unroll
    for (int j = 0; j < 2; j++) {
        v16h B;
#pragma unroll
        for (int i = 0; i < 16; i++)
            B[i] = (_Float16)W[(hf * 16 + i) * 32 + j * 16 + n];
        v8f C = {};
        v8f Dv = __builtin_amdgcn_wmma_f32_16x16x32_f16(
            false, A, false, B, (short)0, C, false, false);
#pragma unroll
        for (int v = 0; v < 8; v++)
            Y[(size_t)(base + v + hf * 8) * 32 + j * 16 + n] = Dv[v];
    }
}

// ---------------------------------------------------------------------------
// Degree / normalization kernels
// ---------------------------------------------------------------------------
__global__ void deg_init(float* deg, int N) {
    int i = blockIdx.x * blockDim.x + threadIdx.x;
    if (i < N) deg[i] = 1.0f;   // self-loop weight
}
__global__ void deg_acc(const long long* __restrict__ ei,
                        const float* __restrict__ ew, float* deg, int E) {
    int e = blockIdx.x * blockDim.x + threadIdx.x;
    if (e < E) unsafeAtomicAdd(&deg[ei[(size_t)E + e]], ew[e]);
}
__global__ void deg_rsqrt(float* deg, int N) {
    int i = blockIdx.x * blockDim.x + threadIdx.x;
    if (i < N) { float d = deg[i]; deg[i] = d > 0.f ? rsqrtf(d) : 0.f; }
}

// out = dis^2 * xw + b   (self-loop term + bias, also serves as init)
__global__ void gcn_selfbias(const float* __restrict__ XW,
                             const float* __restrict__ dis,
                             const float* __restrict__ b,
                             float* __restrict__ OUT, int N) {
    int idx = blockIdx.x * blockDim.x + threadIdx.x;
    if (idx >= N * 32) return;
    int nn = idx >> 5, k = idx & 31;
    float di = dis[nn];
    OUT[idx] = di * di * XW[idx] + b[k];
}

// wave per edge, lane = channel: OUT[dst] += dis[src]*ew*dis[dst] * XW[src]
__global__ void gcn_edge(const long long* __restrict__ ei,
                         const float* __restrict__ ew,
                         const float* __restrict__ dis,
                         const float* __restrict__ XW,
                         float* __restrict__ OUT, int E) {
    int e = blockIdx.x * (blockDim.x >> 5) + (threadIdx.x >> 5);
    int lane = threadIdx.x & 31;
    if (e >= E) return;
    long long s = ei[e], d = ei[(size_t)E + e];
    float nrm = dis[s] * ew[e] * dis[d];
    float val = nrm * XW[(size_t)s * 32 + lane];
    unsafeAtomicAdd(&OUT[(size_t)d * 32 + lane], val);
}

// pred = relu(concat(x2, node_x) @ lin_W + lin_b)
__global__ void final_lin(const float* __restrict__ X2,
                          const float* __restrict__ nx,
                          const float* __restrict__ linW,
                          const float* __restrict__ linb,
                          float* __restrict__ out, int N) {
    int i = blockIdx.x * blockDim.x + threadIdx.x;
    if (i >= N) return;
    float acc = linb[0];
#pragma unroll
    for (int k = 0; k < 32; k++) acc += X2[(size_t)i * 32 + k] * linW[k];
#pragma unroll
    for (int j = 0; j < 4; j++) acc += nx[(size_t)i * 4 + j] * linW[32 + j];
    out[i] = fmaxf(acc, 0.f);
}

// ---------------------------------------------------------------------------
extern "C" void kernel_launch(void* const* d_in, const int* in_sizes, int n_in,
                              void* d_out, int out_size, void* d_ws, size_t ws_size,
                              hipStream_t stream)
{
    const float*     node_x = (const float*)d_in[0];
    const float*     yx     = (const float*)d_in[1];
    const long long* ei     = (const long long*)d_in[2];
    const float*     ew     = (const float*)d_in[3];
    const float*     w_ih   = (const float*)d_in[4];
    const float*     w_hh   = (const float*)d_in[5];
    const float*     b_ih   = (const float*)d_in[6];
    const float*     b_hh   = (const float*)d_in[7];
    const float*     attW   = (const float*)d_in[8];
    const float*     fcW    = (const float*)d_in[9];
    const float*     fcb    = (const float*)d_in[10];
    const float*     g1W    = (const float*)d_in[11];
    const float*     g1b    = (const float*)d_in[12];
    const float*     g2W    = (const float*)d_in[13];
    const float*     g2b    = (const float*)d_in[14];
    const float*     linW   = (const float*)d_in[15];
    const float*     linb   = (const float*)d_in[16];

    const int N = in_sizes[0] / 4;   // node_x is [N,4]
    const int E = in_sizes[3];       // edge_weight is [E]

    const size_t seg = (((size_t)N * 32 * sizeof(float)) + 255) & ~(size_t)255;
    char* ws = (char*)d_ws;
    float* x1b  = (float*)(ws);            // N*32
    float* bufA = (float*)(ws + seg);      // N*32
    float* bufB = (float*)(ws + 2 * seg);  // N*32
    float* dis  = (float*)(ws + 3 * seg);  // N

    const int tiles = (N + 15) / 16;
    const int tn    = (N + 255) / 256;
    const int tnk   = (N * 32 + 255) / 256;
    const int te    = (E + 255) / 256;
    const int tew   = ((E * 32) + 255) / 256;  // wave-per-edge grid

    // 1. LSTM + attention + FC  -> x1b
    lstm_att_fc<<<tiles, 32, 0, stream>>>(yx, w_ih, w_hh, b_ih, b_hh,
                                          attW, fcW, fcb, x1b, N);
    // 2. degree -> D^-1/2
    deg_init <<<tn, 256, 0, stream>>>(dis, N);
    deg_acc  <<<te, 256, 0, stream>>>(ei, ew, dis, E);
    deg_rsqrt<<<tn, 256, 0, stream>>>(dis, N);
    // 3. GCN layer 1: bufB = norm-aggregate(x1b @ g1W) + b1
    gemm_nx32   <<<tiles, 32, 0, stream>>>(x1b, g1W, bufA, N);
    gcn_selfbias<<<tnk, 256, 0, stream>>>(bufA, dis, g1b, bufB, N);
    gcn_edge    <<<tew, 256, 0, stream>>>(ei, ew, dis, bufA, bufB, E);
    // 4. GCN layer 2: bufA = norm-aggregate(bufB @ g2W) + b2
    gemm_nx32   <<<tiles, 32, 0, stream>>>(bufB, g2W, x1b, N);
    gcn_selfbias<<<tnk, 256, 0, stream>>>(x1b, dis, g2b, bufA, N);
    gcn_edge    <<<tew, 256, 0, stream>>>(ei, ew, dis, x1b, bufA, E);
    // 5. final linear + relu
    final_lin<<<(N + 255) / 256, 256, 0, stream>>>(bufA, node_x, linW, linb,
                                                   (float*)d_out, N);
}
